// ModulatedSparseTransformerCrossBlock_77824807404196
// MI455X (gfx1250) — compile-verified, matrix operand order checked by reading, not run on hardware
//
#include <hip/hip_runtime.h>
#include <stdint.h>
#include <stddef.h>

// ---------------------------------------------------------------- types
typedef float  v8f   __attribute__((ext_vector_type(8)));
typedef __bf16 v16bf __attribute__((ext_vector_type(16)));

#define DEVI __device__ __forceinline__

// problem constants
#define NTOK 16384
#define CDIM 1024
#define C6   6144
#define BATCH 4
#define NB   4096
#define HEADS 16
#define DH   64
#define LCTX 256
#define WSZ_ 512

// ---------------------------------------------------------------- helpers
DEVI uint16_t f2bf(float f) {
  uint32_t u = __float_as_uint(f);
  u += 0x7fffu + ((u >> 16) & 1u);      // round-to-nearest-even
  return (uint16_t)(u >> 16);
}

// K-pair base for 16-bit WMMA A/B operand layout (ISA 7.12.2):
// lanes 0-15 hold K {0..7,16..23}, lanes 16-31 hold K {8..15,24..31}
DEVI int kpair(int v, int lane) {
  int kh = (lane >= 16) ? 8 : 0;
  return ((v < 4) ? (2 * v) : (16 + 2 * (v - 4))) + kh;
}

union BfVec { v16bf v; uint32_t u[8]; uint16_t h[16]; };

// operand whose K index is contiguous in memory (A tiles, K tiles, W^T tiles)
DEVI v16bf load_kcontig(const uint16_t* base, int stride) {
  int lane = threadIdx.x & 31;
  const uint16_t* p = base + (size_t)(lane & 15) * (size_t)stride;
  BfVec r;
#pragma unroll
  for (int v = 0; v < 8; ++v) {
    int k0 = kpair(v, lane);
    r.u[v] = *(const uint32_t*)(p + k0);
  }
  return r.v;
}

DEVI v8f wmma_bf16(v16bf a, v16bf b, v8f c) {
  return __builtin_amdgcn_wmma_f32_16x16x32_bf16(false, a, false, b, (short)0, c,
                                                 false, false);
}

// CDNA5 async global->LDS copy (ASYNCcnt path)
DEVI void async_b128(uint32_t lds_off, const void* gaddr) {
  asm volatile("global_load_async_to_lds_b128 %0, %1, off"
               :: "v"(lds_off), "v"(gaddr) : "memory");
}
DEVI void wait_async() { asm volatile("s_wait_asynccnt 0x0" ::: "memory"); }

DEVI float gelu_tanh(float v) {
  float c = v + 0.044715f * v * v * v;
  return 0.5f * v * (1.0f + tanhf(0.7978845608028654f * c));
}

// ---------------------------------------------------------------- fp32 -> bf16 (row-major)
__global__ __launch_bounds__(256) void k_f32_bf16(const float* __restrict__ src,
                                                  uint16_t* __restrict__ dst, int n) {
  int i = blockIdx.x * 256 + threadIdx.x;
  if (i < n) dst[i] = f2bf(src[i]);
}

// ---------------------------------------------------------------- fp32 [K,N] -> bf16 W^T [N,K]
__global__ __launch_bounds__(256) void k_transpose_bf16(const float* __restrict__ src,
                                                        uint16_t* __restrict__ dst,
                                                        int K, int N) {
  __shared__ uint16_t t[32][33];
  int nb = blockIdx.x * 32, kb = blockIdx.y * 32;
  int tx = threadIdx.x & 31, ty = threadIdx.x >> 5;   // 32 x 8
#pragma unroll
  for (int i = 0; i < 32; i += 8)
    t[ty + i][tx] = f2bf(src[(size_t)(kb + ty + i) * N + nb + tx]);
  __syncthreads();
#pragma unroll
  for (int i = 0; i < 32; i += 8)
    dst[(size_t)(nb + ty + i) * K + kb + tx] = t[tx][ty + i];
}

// ---------------------------------------------------------------- m6 = silu(mod)@W_mod + b
__global__ __launch_bounds__(256) void k_mod(const float* __restrict__ mod,
                                             const float* __restrict__ W,
                                             const float* __restrict__ bias,
                                             float* __restrict__ m6) {
  __shared__ float sm[CDIM];
  int b = blockIdx.y;
  for (int i = threadIdx.x; i < CDIM; i += 256) {
    float v = mod[b * CDIM + i];
    sm[i] = v / (1.0f + __expf(-v));
  }
  __syncthreads();
  int col = blockIdx.x * 256 + threadIdx.x;
  float s = 0.0f;
  for (int k = 0; k < CDIM; ++k) s += sm[k] * W[(size_t)k * C6 + col];
  m6[b * C6 + col] = s + bias[col];
}

// ---------------------------------------------------------------- LN + modulate -> bf16
__global__ __launch_bounds__(256) void k_ln(const float* __restrict__ in,
                                            float* __restrict__ copy_out,
                                            uint16_t* __restrict__ out_bf,
                                            const float* __restrict__ scv,
                                            const float* __restrict__ shv,
                                            int vb_stride, float scale_add) {
  __shared__ float rs[256], rq[256];
  int row = blockIdx.x;
  int b   = row / NB;
  const float* x = in + (size_t)row * CDIM;
  float vals[4], s = 0.f, s2 = 0.f;
#pragma unroll
  for (int i = 0; i < 4; ++i) {
    float v = x[threadIdx.x + i * 256];
    vals[i] = v; s += v; s2 += v * v;
  }
  rs[threadIdx.x] = s; rq[threadIdx.x] = s2;
  __syncthreads();
  for (int st = 128; st > 0; st >>= 1) {
    if (threadIdx.x < st) { rs[threadIdx.x] += rs[threadIdx.x + st];
                            rq[threadIdx.x] += rq[threadIdx.x + st]; }
    __syncthreads();
  }
  float mean = rs[0] * (1.0f / CDIM);
  float var  = rq[0] * (1.0f / CDIM) - mean * mean;
  float inv  = rsqrtf(var + 1e-6f);
#pragma unroll
  for (int i = 0; i < 4; ++i) {
    int c = threadIdx.x + i * 256;
    float h = (vals[i] - mean) * inv;
    float sc = scale_add + scv[(size_t)vb_stride * b + c];
    h = h * sc + shv[(size_t)vb_stride * b + c];
    out_bf[(size_t)row * CDIM + c] = f2bf(h);
    if (copy_out) copy_out[(size_t)row * CDIM + c] = vals[i];
  }
}

// ---------------------------------------------------------------- bf16 WMMA GEMM
// C[M,N] = A[M,K](bf16,row-major) * W^T[N,K](bf16) + bias, 128x128 tile, 8 waves.
// Both tiles staged with async global->LDS copies (K-contiguous in both).
// mode 0: store bf16; mode 1: gelu -> bf16; mode 2: outF += gate * val (residual)
__global__ __launch_bounds__(256)
void k_gemm(const uint16_t* __restrict__ A, int lda,
            const uint16_t* __restrict__ WT,   // [N,K]
            int K,
            const float* __restrict__ bias,
            int mode,
            uint16_t* __restrict__ outBf, int ldo,
            float* __restrict__ outF,
            const float* __restrict__ m6, int gate_off) {
  __shared__ uint16_t As[2][128 * 32];   // [m][k]
  __shared__ uint16_t Bs[2][128 * 32];   // [n][k]
  const int tid  = threadIdx.x;
  const int lane = tid & 31;
  const int wave = tid >> 5;
  const int wm = wave & 3;          // 4 waves along M (32 rows each)
  const int wn = wave >> 2;         // 2 waves along N (64 cols each)
  const int n0 = blockIdx.x * 128;
  const int m0 = blockIdx.y * 128;

  v8f acc[2][4];
  v8f z = {0.f, 0.f, 0.f, 0.f, 0.f, 0.f, 0.f, 0.f};
#pragma unroll
  for (int i = 0; i < 2; ++i)
#pragma unroll
    for (int j = 0; j < 4; ++j) acc[i][j] = z;

  // each thread owns chunk (row = tid>>2 [+64], kcol = (tid&3)*8) for both tiles
  const int srow = tid >> 2;
  const int scol = (tid & 3) * 8;
  auto stage = [&](int buf, int kt) {
#pragma unroll
    for (int it = 0; it < 2; ++it) {
      int row = srow + it * 64;
      uint32_t la = (uint32_t)(size_t)&As[buf][row * 32 + scol];
      async_b128(la, A + (size_t)(m0 + row) * lda + kt + scol);
      uint32_t lb = (uint32_t)(size_t)&Bs[buf][row * 32 + scol];
      async_b128(lb, WT + (size_t)(n0 + row) * K + kt + scol);
    }
  };

  stage(0, 0);
  wait_async();
  __syncthreads();

  for (int kt = 0; kt < K; kt += 32) {
    int buf = (kt >> 5) & 1;
    if (kt + 32 < K) stage(buf ^ 1, kt + 32);

    // load all six operands into distinct registers, then 8 back-to-back WMMAs
    v16bf a0 = load_kcontig(&As[buf][(wm * 32 + 0) * 32], 32);
    v16bf a1 = load_kcontig(&As[buf][(wm * 32 + 16) * 32], 32);
    v16bf b0 = load_kcontig(&Bs[buf][(wn * 64 + 0) * 32], 32);
    v16bf b1 = load_kcontig(&Bs[buf][(wn * 64 + 16) * 32], 32);
    v16bf b2 = load_kcontig(&Bs[buf][(wn * 64 + 32) * 32], 32);
    v16bf b3 = load_kcontig(&Bs[buf][(wn * 64 + 48) * 32], 32);
    acc[0][0] = wmma_bf16(a0, b0, acc[0][0]);
    acc[1][0] = wmma_bf16(a1, b0, acc[1][0]);
    acc[0][1] = wmma_bf16(a0, b1, acc[0][1]);
    acc[1][1] = wmma_bf16(a1, b1, acc[1][1]);
    acc[0][2] = wmma_bf16(a0, b2, acc[0][2]);
    acc[1][2] = wmma_bf16(a1, b2, acc[1][2]);
    acc[0][3] = wmma_bf16(a0, b3, acc[0][3]);
    acc[1][3] = wmma_bf16(a1, b3, acc[1][3]);

    wait_async();
    __syncthreads();
  }

  const int rhalf = (lane >= 16) ? 8 : 0;
#pragma unroll
  for (int tm = 0; tm < 2; ++tm)
#pragma unroll
    for (int tn = 0; tn < 4; ++tn) {
      int colb = n0 + wn * 64 + tn * 16 + (lane & 15);
      float bv = bias[colb];
#pragma unroll
      for (int r = 0; r < 8; ++r) {
        int row = m0 + wm * 32 + tm * 16 + r + rhalf;
        float vv = acc[tm][tn][r] + bv;
        if (mode == 0) {
          outBf[(size_t)row * ldo + colb] = f2bf(vv);
        } else if (mode == 1) {
          outBf[(size_t)row * ldo + colb] = f2bf(gelu_tanh(vv));
        } else {
          if (gate_off >= 0) {
            int b = row / NB;
            vv *= m6[(size_t)b * C6 + gate_off + colb];
          }
          size_t idx = (size_t)row * ldo + colb;
          outF[idx] = outF[idx] + vv;
        }
      }
    }
}

// ---------------------------------------------------------------- flash attention
// One workgroup = (group, head, 128-row q chunk); 8 waves x 16 q rows.
__global__ __launch_bounds__(256)
void k_attn(const uint16_t* __restrict__ Q, int q_rs, int q_group_rows, int qchunks,
            const uint16_t* __restrict__ KV, int kv_rs, int kv_group_rows,
            int k_co_base, int v_co_base, int n_keys,
            uint16_t* __restrict__ O) {
  __shared__ uint16_t Ks[32 * 64];      // [key][dh]
  __shared__ uint16_t VsT[64 * 32];     // [dh][key]  (transposed during staging)
  __shared__ uint16_t Ps[8][16 * 32];

  const int tid  = threadIdx.x;
  const int lane = tid & 31;
  const int wave = tid >> 5;
  int bid = blockIdx.x;
  int qc = bid % qchunks;
  int h  = (bid / qchunks) % HEADS;
  int g  = bid / (qchunks * HEADS);

  const int qrow0    = g * q_group_rows + qc * 128 + wave * 16;
  const int key_base = g * kv_group_rows;
  const int qco = h * DH;
  const int kco = k_co_base + h * DH;
  const int vco = v_co_base + h * DH;

  // Q in A-layout registers (16 rows x 64 dh = two 16x32 operands)
  v16bf qa0 = load_kcontig(Q + (size_t)qrow0 * q_rs + qco, q_rs);
  v16bf qa1 = load_kcontig(Q + (size_t)qrow0 * q_rs + qco + 32, q_rs);

  v8f o[4];
  v8f z = {0.f, 0.f, 0.f, 0.f, 0.f, 0.f, 0.f, 0.f};
#pragma unroll
  for (int t = 0; t < 4; ++t) o[t] = z;
  float m_run[8], l_run[8];
#pragma unroll
  for (int r = 0; r < 8; ++r) { m_run[r] = -1e30f; l_run[r] = 0.0f; }

  const int rhalf = (lane >= 16) ? 8 : 0;
  const int skey = tid >> 3;            // 0..31
  const int sd0  = (tid & 7) * 8;       // 0..56

  for (int j = 0; j < n_keys; j += 32) {
    int grow = key_base + j + skey;
    // K tile: async straight copy (K-operand wants [key][dh], dh contiguous)
    async_b128((uint32_t)(size_t)&Ks[skey * 64 + sd0],
               KV + (size_t)grow * kv_rs + kco + sd0);
    // V tile: load b128 + transpose-scatter into [dh][key]
    {
      uint4 d = *(const uint4*)(KV + (size_t)grow * kv_rs + vco + sd0);
      const uint16_t* hv = (const uint16_t*)&d;
#pragma unroll
      for (int i = 0; i < 8; ++i) VsT[(sd0 + i) * 32 + skey] = hv[i];
    }
    wait_async();
    __syncthreads();

    // S = scale * Q K^T  (two 16-key subtiles)
    v8f S[2];
#pragma unroll
    for (int t2 = 0; t2 < 2; ++t2) {
      v16bf kb0 = load_kcontig(&Ks[(t2 * 16) * 64 + 0], 64);
      v16bf kb1 = load_kcontig(&Ks[(t2 * 16) * 64 + 32], 64);
      v8f s = wmma_bf16(qa0, kb0, z);
      s = wmma_bf16(qa1, kb1, s);
      S[t2] = s * 0.125f;               // Dh^-0.5
    }

    // online softmax per row (rows across vgpr index, cols across 16 lanes)
    float alpha[8];
#pragma unroll
    for (int r = 0; r < 8; ++r) {
      float v = fmaxf(S[0][r], S[1][r]);
#pragma unroll
      for (int off = 1; off < 16; off <<= 1) v = fmaxf(v, __shfl_xor(v, off, 32));
      float mnew = fmaxf(m_run[r], v);
      alpha[r] = __expf(m_run[r] - mnew);
      float p0 = __expf(S[0][r] - mnew);
      float p1 = __expf(S[1][r] - mnew);
      S[0][r] = p0; S[1][r] = p1;
      float srow = p0 + p1;
#pragma unroll
      for (int off = 1; off < 16; off <<= 1) srow += __shfl_xor(srow, off, 32);
      l_run[r] = l_run[r] * alpha[r] + srow;
      m_run[r] = mnew;
    }
#pragma unroll
    for (int t = 0; t < 4; ++t)
#pragma unroll
      for (int r = 0; r < 8; ++r) o[t][r] *= alpha[r];

    // relayout P (D layout) -> A layout via per-wave LDS staging
#pragma unroll
    for (int t2 = 0; t2 < 2; ++t2)
#pragma unroll
      for (int r = 0; r < 8; ++r)
        Ps[wave][(r + rhalf) * 32 + t2 * 16 + (lane & 15)] = f2bf(S[t2][r]);

    v16bf pa = load_kcontig(&Ps[wave][0], 32);
    v16bf v0 = load_kcontig(&VsT[(0 * 16) * 32], 32);
    v16bf v1 = load_kcontig(&VsT[(1 * 16) * 32], 32);
    v16bf v2 = load_kcontig(&VsT[(2 * 16) * 32], 32);
    v16bf v3 = load_kcontig(&VsT[(3 * 16) * 32], 32);
    o[0] = wmma_bf16(pa, v0, o[0]);
    o[1] = wmma_bf16(pa, v1, o[1]);
    o[2] = wmma_bf16(pa, v2, o[2]);
    o[3] = wmma_bf16(pa, v3, o[3]);
    __syncthreads();
  }

  float inv_l[8];
#pragma unroll
  for (int r = 0; r < 8; ++r) inv_l[r] = 1.0f / l_run[r];
#pragma unroll
  for (int t = 0; t < 4; ++t)
#pragma unroll
    for (int r = 0; r < 8; ++r) {
      int row = qrow0 + r + rhalf;
      int col = h * DH + t * 16 + (lane & 15);
      O[(size_t)row * CDIM + col] = f2bf(o[t][r] * inv_l[r]);
    }
}

// ---------------------------------------------------------------- launch
static inline size_t alignup(size_t x) { return (x + 255) & ~(size_t)255; }

extern "C" void kernel_launch(void* const* d_in, const int* in_sizes, int n_in,
                              void* d_out, int out_size, void* d_ws, size_t ws_size,
                              hipStream_t stream) {
  (void)in_sizes; (void)n_in; (void)out_size; (void)ws_size;
  const float* x      = (const float*)d_in[0];
  const float* ctx    = (const float*)d_in[1];
  const float* mod    = (const float*)d_in[2];
  const float* W_mod  = (const float*)d_in[3];
  const float* b_mod  = (const float*)d_in[4];
  const float* gamma2 = (const float*)d_in[5];
  const float* beta2  = (const float*)d_in[6];
  const float* W_qkv  = (const float*)d_in[7];
  const float* b_qkv  = (const float*)d_in[8];
  const float* W_so   = (const float*)d_in[9];
  const float* b_so   = (const float*)d_in[10];
  const float* W_q    = (const float*)d_in[11];
  const float* b_q    = (const float*)d_in[12];
  const float* W_kv   = (const float*)d_in[13];
  const float* b_kv   = (const float*)d_in[14];
  const float* W_co   = (const float*)d_in[15];
  const float* b_co   = (const float*)d_in[16];
  const float* W_m1   = (const float*)d_in[17];
  const float* b_m1   = (const float*)d_in[18];
  const float* W_m2   = (const float*)d_in[19];
  const float* b_m2   = (const float*)d_in[20];
  float* out = (float*)d_out;

  char* ws = (char*)d_ws;
  size_t off = 0;
  auto take = [&](size_t bytes) { void* p = ws + off; off += alignup(bytes); return p; };

  float*    m6    = (float*)   take((size_t)BATCH * C6 * 4);
  uint16_t* wqkv  = (uint16_t*)take((size_t)CDIM * 3072 * 2);   // all weights stored W^T [N,K]
  uint16_t* wso   = (uint16_t*)take((size_t)CDIM * CDIM * 2);
  uint16_t* wq    = (uint16_t*)take((size_t)CDIM * CDIM * 2);
  uint16_t* wkv   = (uint16_t*)take((size_t)CDIM * 2048 * 2);
  uint16_t* wco   = (uint16_t*)take((size_t)CDIM * CDIM * 2);
  uint16_t* wm1   = (uint16_t*)take((size_t)CDIM * 4096 * 2);
  uint16_t* wm2   = (uint16_t*)take((size_t)4096 * CDIM * 2);
  uint16_t* ctxb  = (uint16_t*)take((size_t)BATCH * LCTX * CDIM * 2);
  uint16_t* hbf   = (uint16_t*)take((size_t)NTOK * CDIM * 2);   // LN outputs
  uint16_t* bufA  = (uint16_t*)take((size_t)NTOK * 4096 * 2);   // qkv / q / mlp-mid
  uint16_t* bufC  = (uint16_t*)take((size_t)NTOK * CDIM * 2);   // attention outputs
  uint16_t* kvbf  = (uint16_t*)take((size_t)BATCH * LCTX * 2048 * 2);

  auto trans = [&](const float* s, uint16_t* d, int K, int N) {
    k_transpose_bf16<<<dim3(N / 32, K / 32), 256, 0, stream>>>(s, d, K, N);
  };
  trans(W_qkv, wqkv, CDIM, 3072);
  trans(W_so,  wso,  CDIM, CDIM);
  trans(W_q,   wq,   CDIM, CDIM);
  trans(W_kv,  wkv,  CDIM, 2048);
  trans(W_co,  wco,  CDIM, CDIM);
  trans(W_m1,  wm1,  CDIM, 4096);
  trans(W_m2,  wm2,  4096, CDIM);
  k_f32_bf16<<<(BATCH * LCTX * CDIM + 255) / 256, 256, 0, stream>>>(
      ctx, ctxb, BATCH * LCTX * CDIM);

  // m6 = silu(mod) @ W_mod + b_mod
  k_mod<<<dim3(C6 / 256, BATCH), 256, 0, stream>>>(mod, W_mod, b_mod, m6);

  // h = ln(x)*(1+sc_msa)+sh_msa ; also copy x into residual stream (d_out)
  k_ln<<<NTOK, 256, 0, stream>>>(x, out, hbf, m6 + CDIM, m6 + 0, C6, 1.0f);

  // qkv = h @ W_qkv + b  -> bufA (bf16, ldo 3072)
  k_gemm<<<dim3(3072 / 128, NTOK / 128), 256, 0, stream>>>(
      hbf, CDIM, wqkv, CDIM, b_qkv, 0, bufA, 3072, nullptr, nullptr, -1);

  // windowed self attention: groups = B*NW = 32, 512 keys, 4 q-chunks
  k_attn<<<32 * HEADS * 4, 256, 0, stream>>>(
      bufA, 3072, WSZ_, 4, bufA, 3072, WSZ_, CDIM, 2 * CDIM, WSZ_, bufC);

  // xb += g_msa * (o @ W_so + b_so)
  k_gemm<<<dim3(CDIM / 128, NTOK / 128), 256, 0, stream>>>(
      bufC, CDIM, wso, CDIM, b_so, 2, nullptr, CDIM, out, m6, 2048);

  // h = ln(xb)*gamma2 + beta2
  k_ln<<<NTOK, 256, 0, stream>>>(out, nullptr, hbf, gamma2, beta2, 0, 0.0f);

  // q = h @ W_q + b_q -> bufA (ldo 1024)
  k_gemm<<<dim3(CDIM / 128, NTOK / 128), 256, 0, stream>>>(
      hbf, CDIM, wq, CDIM, b_q, 0, bufA, CDIM, nullptr, nullptr, -1);

  // kv = context @ W_kv + b_kv -> kvbf (M = B*L = 1024, N = 2048)
  k_gemm<<<dim3(2048 / 128, (BATCH * LCTX) / 128), 256, 0, stream>>>(
      ctxb, CDIM, wkv, CDIM, b_kv, 0, kvbf, 2048, nullptr, nullptr, -1);

  // cross attention: groups = B = 4, 256 keys, 32 q-chunks
  k_attn<<<BATCH * HEADS * 32, 256, 0, stream>>>(
      bufA, CDIM, NB, 32, kvbf, 2048, LCTX, 0, CDIM, LCTX, bufC);

  // xb += o @ W_co + b_co
  k_gemm<<<dim3(CDIM / 128, NTOK / 128), 256, 0, stream>>>(
      bufC, CDIM, wco, CDIM, b_co, 2, nullptr, CDIM, out, m6, -1);

  // h = ln(xb)*(1+sc_mlp)+sh_mlp
  k_ln<<<NTOK, 256, 0, stream>>>(out, nullptr, hbf, m6 + 4096, m6 + 3072, C6, 1.0f);

  // mid = gelu(h @ W_m1 + b_m1) -> bufA (ldo 4096)
  k_gemm<<<dim3(4096 / 128, NTOK / 128), 256, 0, stream>>>(
      hbf, CDIM, wm1, CDIM, b_m1, 1, bufA, 4096, nullptr, nullptr, -1);

  // xb += g_mlp * (mid @ W_m2 + b_m2)
  k_gemm<<<dim3(CDIM / 128, NTOK / 128), 256, 0, stream>>>(
      bufA, 4096, wm2, 4096, b_m2, 2, nullptr, CDIM, out, m6, 5120);
}
